// cmfsm_75170517614673
// MI455X (gfx1250) — matrix operand order repaired and checked by reading, not compiled, not run on hardware
//
#include <hip/hip_runtime.h>
#include <hip/hip_bf16.h>

typedef __attribute__((ext_vector_type(16))) _Float16 v16h;
typedef __attribute__((ext_vector_type(8)))  float    v8f;

#define HH 512
#define WW 1024
#define HL 256
#define WL 512
#define TILE_X 128
#define ROWS 4
#define HRS 133   // padded LDS row stride (dwords) for hr tile: 16*HRS % 64 != 0
#define LRS 65    // padded LDS row stride for lr tile

// weight LDS offsets in _Float16 elements; each 16x16x32 A-tile = 32 lanes * 16 elems = 512
#define W0_OFF 0
#define W1_OFF (8*4*512)            // 16384
#define W2_OFF (W1_OFF + 4*4*512)   // 24576
#define W3_OFF (W2_OFF + 2*2*512)   // 26624
#define W_TOT  (W3_OFF + 1*512)     // 27136

struct __align__(32) Smem {
  _Float16 w[W_TOT];       // 54272 B, weights in A-operand order
  float hr[32 * HRS];      // 17024 B
  float lr[32 * LRS];      //  8320 B
  float wc[16];            // folded w5 @ w4
  float w2v[4];            // 2x2-periodic weights2 values
};

__device__ __forceinline__ float leaky1(float x) { return x >= 0.f ? x : 0.01f * x; }

__device__ __forceinline__ v8f wmma16(v16h a, v16h b, v8f c) {
  return __builtin_amdgcn_wmma_f32_16x16x32_f16(false, a, false, b, (short)0, c, false, false);
}

// Stage W[Mout,K] (f32, row-major) into LDS as f16 A-operand tiles.
// A layout used here (16-bit A 16x32, contiguous k, not ISA table order): lane l: m=16*mt+(l&15), h=l>>4;
// element e -> k = 32*kb + 16h + e
__device__ __forceinline__ void stage_weights(const float* __restrict__ W, _Float16* dst,
                                              int MT, int KB, int K, int tid) {
  int total = MT * KB * 512;
  for (int i = tid; i < total; i += 256) {
    int e    = i & 15;
    int lane = (i >> 4) & 31;
    int t    = i >> 9;
    int kb   = t % KB;
    int mt   = t / KB;
    int h    = lane >> 4;
    int m    = 16 * mt + (lane & 15);
    int k    = 32 * kb + 16 * h + e;
    dst[i] = (_Float16)W[m * K + k];
  }
}

// One MLP layer: D = leaky(W * Bin), re-emitted as B operands for the next layer.
// B layout used here (contiguous k, same as A): lane l holds column n=l&15, K=16h+e.  A 32-K output
// block = D tiles (2p, 2p+1); lane keeps its own 8 rows and swaps 8 with lane^16.
template<int MT, int KB>
__device__ __forceinline__ void mlp_layer(const v16h* Wt, const v16h* Bin, v16h* Bout, int lane) {
  const int h = lane >> 4;
#pragma unroll
  for (int p = 0; p < MT / 2; ++p) {
    v8f accE = {}; v8f accO = {};
#pragma unroll
    for (int kb = 0; kb < KB; ++kb) {
      v16h aE = Wt[((2 * p) * KB + kb) * 32 + lane];
      v16h aO = Wt[((2 * p + 1) * KB + kb) * 32 + lane];
      accE = wmma16(aE, Bin[kb], accE);
      accO = wmma16(aO, Bin[kb], accO);
    }
    v16h o;
#pragma unroll
    for (int r = 0; r < 8; ++r) {
      float own  = h ? accO[r] : accE[r];   // our half's values of tile 2p+h
      float send = h ? accE[r] : accO[r];   // what the partner lane needs
      float got  = __shfl_xor(send, 16, 32);
      own = leaky1(own); got = leaky1(got);
      o[r]     = (_Float16)(h ? got : own); // element e=r   <=> K=16h+r half
      o[8 + r] = (_Float16)(h ? own : got); // element e=8+r
    }
    Bout[p] = o;
  }
}

__global__ __launch_bounds__(256) void fuse_map_kernel(
    const float* __restrict__ lr, const float* __restrict__ hr,
    const float* __restrict__ w0, const float* __restrict__ w1,
    const float* __restrict__ w2, const float* __restrict__ w3,
    const float* __restrict__ w4, const float* __restrict__ w5,
    const float* __restrict__ s2w0, const float* __restrict__ s2w1,
    const float* __restrict__ s2w2, const float* __restrict__ fuse,
    float* __restrict__ out)
{
  __shared__ Smem sm;
  const int tid  = threadIdx.x;
  const int lane = tid & 31;
  const int wv   = tid >> 5;     // wave 0..7, owns 16 pixels
  const int n    = lane & 15;
  const int h    = lane >> 4;
  const int x0   = blockIdx.x * TILE_X;
  const int y0   = blockIdx.y * ROWS;

  // ---- once per workgroup: stage weights + tiny precomputes ----
  stage_weights(w0, sm.w + W0_OFF, 8, 4, 128, tid);
  stage_weights(w1, sm.w + W1_OFF, 4, 4, 128, tid);
  stage_weights(w2, sm.w + W2_OFF, 2, 2,  64, tid);
  stage_weights(w3, sm.w + W3_OFF, 1, 1,  32, tid);
  if (tid < 16) {              // wc = s1_w5 @ s1_w4 (no activation between them)
    float s = 0.f;
#pragma unroll
    for (int j = 0; j < 8; ++j) s += w5[j] * w4[j * 16 + tid];
    sm.wc[tid] = s;
  }
  if (tid < 4) {               // weights2 is 2x2 periodic: in = (sx, sy, sqrt(2))
    float in0 = (tid & 1) ? 1.f : -1.f;  // x parity channel (d0)
    float in1 = (tid & 2) ? 1.f : -1.f;  // y parity channel (d0.T)
    float in2 = 1.4142135623730951f;
    float a0 = leaky1(s2w0[0]*in0 + s2w0[1]*in1 + s2w0[2]*in2);
    float a1 = leaky1(s2w0[3]*in0 + s2w0[4]*in1 + s2w0[5]*in2);
    float a2 = leaky1(s2w0[6]*in0 + s2w0[7]*in1 + s2w0[8]*in2);
    float b0 = leaky1(s2w1[0]*a0 + s2w1[1]*a1 + s2w1[2]*a2);
    float b1 = leaky1(s2w1[3]*a0 + s2w1[4]*a1 + s2w1[5]*a2);
    sm.w2v[tid] = s2w2[0]*b0 + s2w2[1]*b1;
  }
  const float fw0 = fabsf(fuse[0]);
  const float fw1 = fabsf(fuse[1]);
  __syncthreads();

  const v16h* Wt0 = (const v16h*)(sm.w + W0_OFF);
  const v16h* Wt1 = (const v16h*)(sm.w + W1_OFF);
  const v16h* Wt2 = (const v16h*)(sm.w + W2_OFF);
  const v16h* Wt3 = (const v16h*)(sm.w + W3_OFF);

  for (int ry = 0; ry < ROWS; ++ry) {
    const int y = y0 + ry;
    // coalesced stage of hr[32][128] and lr[32][64] tiles for this row
    for (int i = tid; i < 32 * TILE_X; i += 256) {
      int c = i >> 7, x = i & 127;
      sm.hr[c * HRS + x] = hr[(size_t)c * (HH * WW) + (size_t)y * WW + (x0 + x)];
    }
    {
      const int yl = y >> 1, xl0 = x0 >> 1;
      for (int i = tid; i < 32 * 64; i += 256) {
        int c = i >> 6, x = i & 63;
        sm.lr[c * LRS + x] = lr[(size_t)c * (HL * WL) + (size_t)yl * WL + (xl0 + x)];
      }
    }
    __syncthreads();

    // Build rep B operands: 4 K-blocks of 32 channels: [lr_up, hr, lr*hr, (lr-hr)^2]
    const int nl = wv * 16 + n;          // local pixel column 0..127
    v16h repB[4];
#pragma unroll
    for (int e = 0; e < 16; ++e) {
      int c = 16 * h + e;                // B layout: K = 16h + e within 32-block
      float lv = sm.lr[c * LRS + (nl >> 1)];
      float hv = sm.hr[c * HRS + nl];
      repB[0][e] = (_Float16)lv;
      repB[1][e] = (_Float16)hv;
      repB[2][e] = (_Float16)(lv * hv);
      float d = lv - hv;
      repB[3][e] = (_Float16)(d * d);
    }

    // MLP chain entirely in registers: 128->128->64->32 (leaky each)
    v16h y0B[4], y1B[2], y2B[1];
    mlp_layer<8, 4>(Wt0, repB, y0B, lane);
    mlp_layer<4, 4>(Wt1, y0B, y1B, lane);
    mlp_layer<2, 2>(Wt2, y1B, y2B, lane);

    // Layer 3 (32->16, leaky) then folded 16->1 dot with wc
    v8f acc = {};
    v16h a3 = Wt3[lane];
    acc = wmma16(a3, y2B[0], acc);
    float part = 0.f;
#pragma unroll
    for (int r = 0; r < 8; ++r) part += sm.wc[8 * h + r] * leaky1(acc[r]);
    float tot = part + __shfl_xor(part, 16, 32);

    if (h == 0) {
      const int x = x0 + nl;
      float wv2 = sm.w2v[((y & 1) << 1) | (x & 1)];
      out[(size_t)y * WW + x] = fw0 * tot + fw1 * wv2;
    }
    __syncthreads();
  }
}

extern "C" void kernel_launch(void* const* d_in, const int* in_sizes, int n_in,
                              void* d_out, int out_size, void* d_ws, size_t ws_size,
                              hipStream_t stream) {
  (void)in_sizes; (void)n_in; (void)out_size; (void)d_ws; (void)ws_size;
  const float* lr   = (const float*)d_in[0];
  const float* hr   = (const float*)d_in[1];
  const float* w0   = (const float*)d_in[2];
  const float* w1   = (const float*)d_in[3];
  const float* w2   = (const float*)d_in[4];
  const float* w3   = (const float*)d_in[5];
  const float* w4   = (const float*)d_in[6];
  const float* w5   = (const float*)d_in[7];
  const float* s2w0 = (const float*)d_in[8];
  const float* s2w1 = (const float*)d_in[9];
  const float* s2w2 = (const float*)d_in[10];
  const float* fuse = (const float*)d_in[11];
  dim3 grid(WW / TILE_X, HH / ROWS);
  fuse_map_kernel<<<grid, dim3(256), 0, stream>>>(lr, hr, w0, w1, w2, w3, w4, w5,
                                                  s2w0, s2w1, s2w2, fuse, (float*)d_out);
}
